// MultiScaleDeformableAttention_23811298689520
// MI455X (gfx1250) — matrix operand order repaired and checked by reading, not compiled, not run on hardware
//
#include <hip/hip_runtime.h>
#include <hip/hip_bf16.h>
#include <cstdint>
#include <cstddef>

// ---------------------------------------------------------------------------
// Multi-scale deformable attention, MI455X (gfx1250, wave32, WMMA).
//
//   K1 gemm<VPROJ>: v = value @ vpk + vpb, masked, transposed to
//                   vt[b*8+h][LEN_V][32]   (32 ch == wave32 lanes -> 128B lines)
//   K2 gemm<LOC>  : off = query @ sok + sob, fused with ref_points -> loc
//   K3 gemm<PLAIN>: logits = query @ ak + ab
//   K4 sample     : softmax(16) + bilinear gather (L2-resident) -> xbuf
//   K5 gemm<PLAIN>: out = xbuf @ okern + out_bias
//
// GEMM: 128 thr = 4 waves, block tile 16x64, one 16x16 WMMA tile per wave,
// K=256 in 8 steps of 32. f32 -> f16 staged through LDS with B swizzled into
// fragment-major layout so every fragment load is ds_load_b128. Global loads
// are float4 (global_load_b128) and software-pipelined one K-step ahead.
// ---------------------------------------------------------------------------

typedef __attribute__((ext_vector_type(16))) _Float16 v16h;
typedef __attribute__((ext_vector_type(8)))  _Float16 v8h;
typedef __attribute__((ext_vector_type(8)))  float    v8f;

#define BSZ     2
#define EMBED   256
#define HEADS   8
#define LEVELS  4
#define POINTS  4
#define NEMBED  32
#define LEN_V   18360
#define LEN_Q   18360
#define KDIM    256

#define MODE_PLAIN 0
#define MODE_VPROJ 1
#define MODE_LOC   2

template <int MODE, int N>
__global__ __launch_bounds__(128) void gemm_wmma(
    const float* __restrict__ A,     // [M][256] row-major f32
    const float* __restrict__ Bm,    // [256][N] row-major f32
    const float* __restrict__ bias,  // [N]
    float* __restrict__ out,
    const float* __restrict__ refp,         // MODE_LOC: ref_points [B][LEN_Q][4][2]
    const unsigned char* __restrict__ mask) // MODE_VPROJ: pad_mask [B][LEN_V]
{
  // A tile row-major (fragment rows are two contiguous 16B chunks);
  // B tile fragment-major: Bf[wv*32 + lane][j] holds the 16 halves lane j-needs.
  __shared__ __align__(16) _Float16 As[16][32];
  __shared__ __align__(16) _Float16 Bf[128][16];

  const int t    = threadIdx.x;
  const int lane = t & 31;
  const int wv   = t >> 5;      // wave 0..3 -> 16-col slice
  const int h16  = lane >> 4;   // K-half selector
  const int mr   = lane & 15;

  const int row0 = blockIdx.x * 16;   // M = 36720 divisible by 16
  const int col0 = blockIdx.y * 64;   // N in {256,128} divisible by 64

  // --- staging maps (fixed trip, fully coalesced) ---
  // A: 16x32 f32 = 128 float4 -> exactly one float4 per thread.
  const int ar  = t >> 3;             // row 0..15
  const int ac4 = t & 7;              // float4 col 0..7
  const float* aptr = A + (size_t)(row0 + ar) * KDIM + ac4 * 4;
  // B: 32x64 f32 = 512 float4 -> 4 float4 per thread; f = t + it*128,
  // bk = f>>4 (k row), bc4 = f&15 (float4 along n) -> b128 loads coalesced.
  const float* bbase = Bm + col0;

  float4 areg;
  float4 breg[4];

  auto load_tiles = [&](int k0) {
    areg = *(const float4*)(aptr + k0);
#pragma unroll
    for (int it = 0; it < 4; ++it) {
      const int f = t + it * 128;
      const int bk = f >> 4, bc4 = f & 15;
      breg[it] = *(const float4*)(bbase + (size_t)(k0 + bk) * N + bc4 * 4);
    }
  };

  auto store_tiles = [&]() {
    // A: pack 4 halves, single 8-byte LDS store (8B aligned).
    union { _Float16 h[4]; uint2 u; } ap;
    ap.h[0] = (_Float16)areg.x; ap.h[1] = (_Float16)areg.y;
    ap.h[2] = (_Float16)areg.z; ap.h[3] = (_Float16)areg.w;
    *reinterpret_cast<uint2*>(&As[ar][ac4 * 4]) = ap.u;
    // B: swizzle-on-store into fragment-major layout.
#pragma unroll
    for (int it = 0; it < 4; ++it) {
      const int f = t + it * 128;
      const int bk = f >> 4, bc4 = f & 15;
      const int j = bk & 15;
      const int lbase = (bk >> 4) << 4;
      const float vals[4] = {breg[it].x, breg[it].y, breg[it].z, breg[it].w};
#pragma unroll
      for (int e = 0; e < 4; ++e) {
        const int n = bc4 * 4 + e;   // tile-local col 0..63
        Bf[(n >> 4) * 32 + lbase + (n & 15)][j] = (_Float16)vals[e];
      }
    }
  };

  v8f acc = {};
  load_tiles(0);

#pragma unroll
  for (int k0 = 0; k0 < KDIM; k0 += 32) {
    store_tiles();
    __syncthreads();
    if (k0 + 32 < KDIM) load_tiles(k0 + 32);  // overlap VMEM with WMMA

    // Fragments: A = two contiguous v8h rows; B = one contiguous 32B row.
    union V16 { v16h v; v8h h[2]; } a, b;
    const v8h* arow = reinterpret_cast<const v8h*>(&As[mr][0]);
    a.h[0] = arow[2 * h16];      // halves 0..7  : K = h16*16 + i
    a.h[1] = arow[2 * h16 + 1];  // halves 8..15 : K = h16*16 + 8 + i
    const v8h* brow = reinterpret_cast<const v8h*>(&Bf[wv * 32 + lane][0]);
    b.h[0] = brow[0];
    b.h[1] = brow[1];

    acc = __builtin_amdgcn_wmma_f32_16x16x32_f16(
        /*neg_a=*/false, a.v, /*neg_b=*/false, b.v,
        /*c_mod=*/(short)0, acc, /*reuse_a=*/false, /*reuse_b=*/false);
    __syncthreads();
  }

  // ---- epilogue: c[r] at (M = r + 8*h16, N = mr) of this wave's tile ----
  const int col = col0 + wv * 16 + mr;
  const float bv = bias[col];
  constexpr float normw[4] = {144.f, 72.f, 36.f, 18.f};
  constexpr float normh[4] = {96.f,  48.f, 24.f, 12.f};
#pragma unroll
  for (int r = 0; r < 8; ++r) {
    const int row = row0 + r + 8 * h16;
    float v = acc[r] + bv;
    if (MODE == MODE_PLAIN) {
      out[(size_t)row * N + col] = v;
    } else if (MODE == MODE_VPROJ) {
      const int b_ = row / LEN_V, vv = row - b_ * LEN_V;
      const int h = col >> 5, d = col & 31;
      if (!mask[(size_t)b_ * LEN_V + vv]) v = 0.f;
      out[(((size_t)(b_ * HEADS + h)) * LEN_V + vv) * NEMBED + d] = v;
    } else { // MODE_LOC: col = h*32 + l*8 + p*2 + c
      const int b_ = row / LEN_Q, q = row - b_ * LEN_Q;
      const int h = col >> 5;
      const int rem = col & 31;
      const int l = rem >> 3;
      const int cc = col & 1;
      const float nrm = cc ? normh[l] : normw[l];
      const float rp = refp[(((size_t)(b_ * LEN_Q + q)) * LEVELS + l) * 2 + cc];
      out[((size_t)(b_ * HEADS + h) * LEN_Q + q) * 32 + rem] = rp + v / nrm;
    }
  }
}

// ---------------------------------------------------------------------------
// Sampling core: one wave per (b, h, q); lane == channel d (NEMBED==32==wave32).
// Each corner fetch is one coalesced 128B line; vt (37.6MB) fits in 192MB L2.
// ---------------------------------------------------------------------------
__global__ __launch_bounds__(256) void msda_sample(
    const float* __restrict__ vt,      // [16][LEN_V][32]
    const float* __restrict__ loc,     // [16][LEN_Q][16][2]
    const float* __restrict__ logits,  // [36720][128] col = h*16 + (l*4+p)
    float* __restrict__ xbuf)          // [36720][256] col = h*32 + d
{
  const int wave = (blockIdx.x * blockDim.x + threadIdx.x) >> 5;
  const int lane = threadIdx.x & 31;
  const int bh = wave / LEN_Q;           // 0..15
  const int q  = wave - bh * LEN_Q;
  const int b  = bh >> 3, h = bh & 7;

  const float* lg = logits + (size_t)(b * LEN_Q + q) * (HEADS * 16) + h * 16;
  float w16[16];
  float mx = -3.402823466e38f;
#pragma unroll
  for (int i = 0; i < 16; ++i) { w16[i] = lg[i]; mx = fmaxf(mx, w16[i]); }
  float s = 0.f;
#pragma unroll
  for (int i = 0; i < 16; ++i) { w16[i] = __expf(w16[i] - mx); s += w16[i]; }
  const float inv = 1.f / s;

  const float* lc = loc + ((size_t)bh * LEN_Q + q) * 32;
  const float* vb = vt + (size_t)bh * LEN_V * NEMBED;

  constexpr int lw[4]  = {144, 72, 36, 18};
  constexpr int lh[4]  = {96,  48, 24, 12};
  constexpr int lst[4] = {0, 13824, 17280, 18144};

  float acc = 0.f;
#pragma unroll
  for (int l = 0; l < LEVELS; ++l) {
    const int W = lw[l], H = lh[l];
    const float* vl = vb + (size_t)lst[l] * NEMBED;
#pragma unroll
    for (int p = 0; p < POINTS; ++p) {
      const float gx = lc[(l * 4 + p) * 2 + 0];
      const float gy = lc[(l * 4 + p) * 2 + 1];
      const float x = gx * (float)W - 0.5f;
      const float y = gy * (float)H - 0.5f;
      const float xf = floorf(x), yf = floorf(y);
      const int x0 = (int)xf, y0 = (int)yf;
      const float fx = x - xf, fy = y - yf;
      const bool ix0 = (x0 >= 0) && (x0 <= W - 1);
      const bool ix1 = (x0 + 1 >= 0) && (x0 + 1 <= W - 1);
      const bool iy0 = (y0 >= 0) && (y0 <= H - 1);
      const bool iy1 = (y0 + 1 >= 0) && (y0 + 1 <= H - 1);
      const float aw = w16[l * 4 + p] * inv;
      const float v00 = (ix0 && iy0) ? vl[((size_t)y0 * W + x0) * NEMBED + lane] : 0.f;
      const float v01 = (ix0 && iy1) ? vl[((size_t)(y0 + 1) * W + x0) * NEMBED + lane] : 0.f;
      const float v10 = (ix1 && iy0) ? vl[((size_t)y0 * W + (x0 + 1)) * NEMBED + lane] : 0.f;
      const float v11 = (ix1 && iy1) ? vl[((size_t)(y0 + 1) * W + (x0 + 1)) * NEMBED + lane] : 0.f;
      acc += aw * ((1.f - fx) * (1.f - fy) * v00 + (1.f - fx) * fy * v01 +
                   fx * (1.f - fy) * v10 + fx * fy * v11);
    }
  }
  xbuf[(size_t)(b * LEN_Q + q) * EMBED + h * NEMBED + lane] = acc;
}

// ---------------------------------------------------------------------------
extern "C" void kernel_launch(void* const* d_in, const int* in_sizes, int n_in,
                              void* d_out, int out_size, void* d_ws, size_t ws_size,
                              hipStream_t stream) {
  (void)in_sizes; (void)n_in; (void)out_size; (void)ws_size;
  const float* query = (const float*)d_in[0];   // [2][18360][256]
  const float* refp  = (const float*)d_in[1];   // [2][18360][4][2]
  const float* value = (const float*)d_in[2];   // [2][18360][256]
  const unsigned char* mask = (const unsigned char*)d_in[3]; // bool [2][18360]
  const float* vpk   = (const float*)d_in[4];   // [256][256] (col = h*32+d)
  const float* vpb   = (const float*)d_in[5];   // [256]
  const float* sok   = (const float*)d_in[6];   // [256][256] (col = h*32+l*8+p*2+c)
  const float* sob   = (const float*)d_in[7];   // [256]
  const float* ak    = (const float*)d_in[8];   // [256][128] (col = h*16+m)
  const float* ab    = (const float*)d_in[9];   // [128]
  const float* okern = (const float*)d_in[10];  // [256][256] (row = h*32+e)
  const float* obias = (const float*)d_in[11];  // [256]
  float* out = (float*)d_out;                   // [2][18360][256]

  float* vt     = (float*)d_ws;                                  // 16*18360*32
  float* loc    = vt     + (size_t)16 * LEN_V * NEMBED;          // 16*18360*32
  float* logits = loc    + (size_t)16 * LEN_Q * 32;              // 36720*128
  float* xbuf   = logits + (size_t)BSZ * LEN_Q * (HEADS * 16);   // 36720*256

  const int MT = (BSZ * LEN_Q) / 16;  // 2295 row tiles
  const dim3 blk(128);

  hipLaunchKernelGGL((gemm_wmma<MODE_VPROJ, 256>), dim3(MT, 4), blk, 0, stream,
                     value, vpk, vpb, vt, nullptr, mask);
  hipLaunchKernelGGL((gemm_wmma<MODE_LOC, 256>), dim3(MT, 4), blk, 0, stream,
                     query, sok, sob, loc, refp, nullptr);
  hipLaunchKernelGGL((gemm_wmma<MODE_PLAIN, 128>), dim3(MT, 2), blk, 0, stream,
                     query, ak, ab, logits, nullptr, nullptr);

  const int nwaves = 16 * LEN_Q;  // 293760 waves, 8 per 256-thread block
  hipLaunchKernelGGL(msda_sample, dim3(nwaves / 8), dim3(256), 0, stream,
                     vt, loc, logits, xbuf);

  hipLaunchKernelGGL((gemm_wmma<MODE_PLAIN, 256>), dim3(MT, 4), blk, 0, stream,
                     xbuf, okern, obias, out, nullptr, nullptr);
}
